// GNNLayer_32736240730704
// MI455X (gfx1250) — matrix operand order repaired and checked by reading, compiled, not run on hardware
//
#include <hip/hip_runtime.h>
#include <stdint.h>

#define N_NODES 50000
#define N_EDGES 800000
#define DIM     128
#define TILE    128
#define RS      68      // LDS row stride in dwords (64 bf16-pairs + 4 pad, 16B aligned)
#define RSH     136     // same stride in halfwords

typedef __attribute__((ext_vector_type(16))) __bf16 v16bf;
typedef __attribute__((ext_vector_type(8)))  float  v8f;
typedef __attribute__((ext_vector_type(4)))  unsigned v4u;
typedef __attribute__((ext_vector_type(8)))  int    v8i_t;
typedef __attribute__((ext_vector_type(4)))  int    v4i_t;

union Frag {
    v16bf v;
    uint4 q[2];
};

static __device__ __forceinline__ unsigned pack2(float a, float b) {
    __bf16 x = (__bf16)a, y = (__bf16)b;
    unsigned short ux = __builtin_bit_cast(unsigned short, x);
    unsigned short uy = __builtin_bit_cast(unsigned short, y);
    return ((unsigned)uy << 16) | (unsigned)ux;
}
static __device__ __forceinline__ unsigned short bf16b(float f) {
    __bf16 h = (__bf16)f;
    return __builtin_bit_cast(unsigned short, h);
}
static __device__ __forceinline__ float bf2f(unsigned short u) {
    return __builtin_bit_cast(float, ((unsigned)u) << 16);
}

// ---- Tensor Data Mover: async DMA of one packed 32KB weight (8192 dwords) to LDS.
// D# per CDNA5 ISA ch.8: flat 1-row tile, data_size=4B, tensor/tile dim0=8192.
static __device__ __forceinline__ void tdm_load_32k(const unsigned* __restrict__ g, unsigned* lds) {
    unsigned long long ga = (unsigned long long)(uintptr_t)g;
    unsigned ldso = (unsigned)(uintptr_t)lds;           // low 32 bits of generic addr = LDS byte offset
    v4u g0 = { 1u,                                       // count=1 (valid descriptor)
               ldso,                                     // lds_addr [63:32]
               (unsigned)ga,                             // global_addr [95:64]
               ((unsigned)(ga >> 32) & 0x01ffffffu) | (2u << 30) };  // global_addr[56:32] | type=2
    v8i_t g1 = { (int)(2u << 16),        // workgroup_mask=0 | data_size=2 (4B)
                 (int)(8192u << 16),     // atomic_barrier_addr=0 | tensor_dim0.lo16 = 8192
                 (int)(1u << 16),        // tensor_dim0.hi16=0    | tensor_dim1.lo16 = 1
                 (int)(8192u << 16),     // tensor_dim1.hi16=0    | tile_dim0 = 8192
                 1,                      // tile_dim1 = 1         | tile_dim2 = 0
                 8192,                   // tensor_dim0_stride.lo32 = 8192
                 0, 0 };
    v4i_t z4 = { 0, 0, 0, 0 };
    v8i_t z8 = { 0, 0, 0, 0, 0, 0, 0, 0 };
    __builtin_amdgcn_tensor_load_to_lds(g0, g1, z4, z4, z8, 0);
}

// All waves: wait for the in-flight TDM (no-op for waves that issued none), then sync.
static __device__ __forceinline__ void wait_w_ready() {
    __builtin_amdgcn_s_wait_tensorcnt(0);
    __syncthreads();
}

// acc[nt] += Act(16 rows of this wave, K=128 bf16) x W(128x128 pre-packed in sW)
static __device__ __forceinline__ void gemm_tile(v8f* acc, const unsigned* act,
                                                 const unsigned* sW, int wave, int lane) {
    const int m   = wave * 16 + (lane & 15);
    const int sel = (lane < 16) ? 0 : 4;   // A-fragment K-pair selector per ISA layout
    for (int kc = 0; kc < 4; ++kc) {
        Frag a;
        const unsigned* ap = act + m * RS + kc * 16;
        a.q[0] = *(const uint4*)(ap + sel);
        a.q[1] = *(const uint4*)(ap + 8 + sel);
#pragma unroll
        for (int nt = 0; nt < 8; ++nt) {
            Frag b;
            const unsigned* bp = sW + (((nt << 2) + kc) * 32 + lane) * 8;
            b.q[0] = ((const uint4*)bp)[0];
            b.q[1] = ((const uint4*)bp)[1];
            acc[nt] = __builtin_amdgcn_wmma_f32_16x16x32_bf16(
                false, a.v, false, b.v, (short)0, acc[nt], false, false);
        }
    }
}

// acc + bias -> (relu) -> bf16 into act rows of this wave; acc reset to 0.
static __device__ __forceinline__ void bias_act_store(v8f* acc, unsigned* act,
                                                      const float* __restrict__ bias,
                                                      int wave, int lane, bool relu) {
    unsigned short* a16 = (unsigned short*)act;
#pragma unroll
    for (int nt = 0; nt < 8; ++nt) {
        int n = nt * 16 + (lane & 15);
        float b = bias[n];
#pragma unroll
        for (int r = 0; r < 8; ++r) {
            int m = wave * 16 + ((lane < 16) ? r : r + 8);
            float v = acc[nt][r] + b;
            if (relu) v = v > 0.f ? v : 0.f;
            a16[m * RSH + n] = bf16b(v);
            acc[nt][r] = 0.f;
        }
    }
}

// Pre-pack one fp32 [out][in] weight into the per-lane bf16 B-fragment layout:
// linear index i = ((nt*4+kc)*32 + lane)*8 + r ; value = {W[o][k], W[o][k+1]}
__global__ void pack_w_kernel(const float* __restrict__ w, unsigned* __restrict__ out) {
    for (int i = threadIdx.x; i < 8192; i += 256) {
        int r    = i & 7;
        int lane = (i >> 3) & 31;
        int kc   = (i >> 8) & 3;
        int nt   = (i >> 10) & 7;
        int o = nt * 16 + (lane & 15);
        int k = kc * 32 + ((r & 4) ? 16 : 0) + ((lane & 16) ? 8 : 0) + 2 * (r & 3);
        out[i] = pack2(w[o * DIM + k], w[o * DIM + k + 1]);
    }
}

__global__ __launch_bounds__(256) void gnn_edge_kernel(
    const float* __restrict__ node_feat, const float* __restrict__ edge_feat,
    const int* __restrict__ src_idx, const int* __restrict__ dst_idx,
    const unsigned* __restrict__ wpack,
    const float* __restrict__ bsrc, const float* __restrict__ bdst,
    const float* __restrict__ bphi1, const float* __restrict__ bphi2,
    const float* __restrict__ bphi3, float* __restrict__ agg) {
    __shared__ unsigned sXs[TILE * RS];   // gathered h_src (bf16), kept for u_mul_e
    __shared__ unsigned sXa[TILE * RS];   // h_dst, then the running activation
    __shared__ unsigned sW0[8192];        // double-buffered packed weight (TDM target)
    __shared__ unsigned sW1[8192];

    const int t = threadIdx.x;
    const int wave = t >> 5, lane = t & 31;
    const long tile = (long)blockIdx.x * TILE;

    // Prefetch weight 0 (Wsrc) via TDM while we do the gathers.
    if (wave == 0) tdm_load_32k(wpack + 0 * 8192, sW0);

    // Stage gathered h_src / h_dst -> bf16 LDS (each thread does half a row; rows are wave-private)
    {
        int row = t >> 1;
        int half = (t & 1) * 64;
        long e = tile + row;
        int si = src_idx[e];
        int di = dst_idx[e];
        const float4* ps = (const float4*)(node_feat + (long)si * DIM + half);
        const float4* pd = (const float4*)(node_feat + (long)di * DIM + half);
        unsigned* xs = sXs + row * RS + (half >> 1);
        unsigned* xa = sXa + row * RS + (half >> 1);
#pragma unroll
        for (int i = 0; i < 16; ++i) {
            float4 v = ps[i];
            xs[2 * i]     = pack2(v.x, v.y);
            xs[2 * i + 1] = pack2(v.z, v.w);
            float4 w = pd[i];
            xa[2 * i]     = pack2(w.x, w.y);
            xa[2 * i + 1] = pack2(w.z, w.w);
        }
    }

    v8f acc[8];
#pragma unroll
    for (int i = 0; i < 8; ++i) acc[i] = (v8f){0.f, 0.f, 0.f, 0.f, 0.f, 0.f, 0.f, 0.f};

    wait_w_ready();                                   // sW0 = Wsrc ready

    // ---- stage 0: h_src @ Wsrc^T  (prefetch Wdst -> sW1)
    if (wave == 0) tdm_load_32k(wpack + 1 * 8192, sW1);
    gemm_tile(acc, sXs, sW0, wave, lane);
    wait_w_ready();

    // ---- stage 1: += h_dst @ Wdst^T (prefetch Wphi1 -> sW0)
    if (wave == 0) tdm_load_32k(wpack + 2 * 8192, sW0);
    gemm_tile(acc, sXa, sW1, wave, lane);
    {   // t = acc + edge_feat + bsrc + bdst ; ReLU (phi is activation-first) -> act
        unsigned short* a16 = (unsigned short*)sXa;
#pragma unroll
        for (int nt = 0; nt < 8; ++nt) {
            int n = nt * 16 + (lane & 15);
            float b2 = bsrc[n] + bdst[n];
#pragma unroll
            for (int r = 0; r < 8; ++r) {
                int m = wave * 16 + ((lane < 16) ? r : r + 8);
                float v = acc[nt][r] + edge_feat[(tile + m) * DIM + n] + b2;
                v = v > 0.f ? v : 0.f;
                a16[m * RSH + n] = bf16b(v);
                acc[nt][r] = 0.f;
            }
        }
    }
    wait_w_ready();

    // ---- stage 2: phi L1 (prefetch Wphi2 -> sW1)
    if (wave == 0) tdm_load_32k(wpack + 3 * 8192, sW1);
    gemm_tile(acc, sXa, sW0, wave, lane);
    bias_act_store(acc, sXa, bphi1, wave, lane, true);
    wait_w_ready();

    // ---- stage 3: phi L2 (prefetch Wphi3 -> sW0)
    if (wave == 0) tdm_load_32k(wpack + 4 * 8192, sW0);
    gemm_tile(acc, sXa, sW1, wave, lane);
    bias_act_store(acc, sXa, bphi2, wave, lane, true);
    wait_w_ready();

    // ---- stage 4: phi L3 (no relu) ; message = h_src * e_emb ; scatter-add
    gemm_tile(acc, sXa, sW0, wave, lane);
    {
        float bb[8];
#pragma unroll
        for (int nt = 0; nt < 8; ++nt) bb[nt] = bphi3[nt * 16 + (lane & 15)];
        const unsigned short* s16 = (const unsigned short*)sXs;
#pragma unroll
        for (int r = 0; r < 8; ++r) {
            int m = wave * 16 + ((lane < 16) ? r : r + 8);
            int dr = dst_idx[tile + m];
            float* aggrow = agg + (long)dr * DIM;
#pragma unroll
            for (int nt = 0; nt < 8; ++nt) {
                int n = nt * 16 + (lane & 15);
                float e = acc[nt][r] + bb[nt];
                float msg = bf2f(s16[m * RSH + n]) * e;
                unsafeAtomicAdd(aggrow + n, msg);
            }
        }
    }
}

__global__ __launch_bounds__(256) void gnn_node_kernel(
    const float* __restrict__ node_feat, const float* __restrict__ agg,
    const unsigned* __restrict__ wpack,
    const float* __restrict__ bpd, const float* __restrict__ bpu,
    const float* __restrict__ bth1, const float* __restrict__ bth2,
    float* __restrict__ out) {
    __shared__ unsigned sXa[TILE * RS];
    __shared__ unsigned sW0[8192];
    __shared__ unsigned sW1[8192];

    const int t = threadIdx.x;
    const int wave = t >> 5, lane = t & 31;
    const long tile = (long)blockIdx.x * TILE;

    // Prefetch Wpd while staging h.
    if (wave == 0) tdm_load_32k(wpack + 7 * 8192, sW0);

    const int row = t >> 1;
    const int half = (t & 1) * 64;
    long rg = tile + row;
    if (rg >= N_NODES) rg = N_NODES - 1;   // clamp staging (results masked on store)
    unsigned* xrow = sXa + row * RS + (half >> 1);

    {   // Stage h
        const float4* p = (const float4*)(node_feat + rg * DIM + half);
#pragma unroll
        for (int i = 0; i < 16; ++i) {
            float4 v = p[i];
            xrow[2 * i]     = pack2(v.x, v.y);
            xrow[2 * i + 1] = pack2(v.z, v.w);
        }
    }

    v8f acc[8];
#pragma unroll
    for (int i = 0; i < 8; ++i) acc[i] = (v8f){0.f, 0.f, 0.f, 0.f, 0.f, 0.f, 0.f, 0.f};

    wait_w_ready();                                    // sW0 = Wpd ready

    // ---- stage 0: h @ Wpd^T (prefetch Wpu -> sW1)
    if (wave == 0) tdm_load_32k(wpack + 8 * 8192, sW1);
    gemm_tile(acc, sXa, sW0, wave, lane);
    {   // Restage agg into sXa (rows are wave-private; own wave's GEMM reads are done)
        const float4* p = (const float4*)(agg + rg * DIM + half);
#pragma unroll
        for (int i = 0; i < 16; ++i) {
            float4 v = p[i];
            xrow[2 * i]     = pack2(v.x, v.y);
            xrow[2 * i + 1] = pack2(v.z, v.w);
        }
    }
    wait_w_ready();

    // ---- stage 1: += agg @ Wpu^T (prefetch Wth1 -> sW0)
    if (wave == 0) tdm_load_32k(wpack + 5 * 8192, sW0);
    gemm_tile(acc, sXa, sW1, wave, lane);
    {   // y = relu(pd + pu + biases) -> act
        unsigned short* a16 = (unsigned short*)sXa;
#pragma unroll
        for (int nt = 0; nt < 8; ++nt) {
            int n = nt * 16 + (lane & 15);
            float b2 = bpd[n] + bpu[n];
#pragma unroll
            for (int r = 0; r < 8; ++r) {
                int m = wave * 16 + ((lane < 16) ? r : r + 8);
                float v = acc[nt][r] + b2;
                v = v > 0.f ? v : 0.f;
                a16[m * RSH + n] = bf16b(v);
                acc[nt][r] = 0.f;
            }
        }
    }
    wait_w_ready();

    // ---- stage 2: theta L1 (prefetch Wth2 -> sW1)
    if (wave == 0) tdm_load_32k(wpack + 6 * 8192, sW1);
    gemm_tile(acc, sXa, sW0, wave, lane);
    bias_act_store(acc, sXa, bth1, wave, lane, true);
    wait_w_ready();

    // ---- stage 3: theta L2 -> out (fp32, bounds-masked)
    gemm_tile(acc, sXa, sW1, wave, lane);
#pragma unroll
    for (int nt = 0; nt < 8; ++nt) {
        int n = nt * 16 + (lane & 15);
        float b = bth2[n];
#pragma unroll
        for (int r = 0; r < 8; ++r) {
            int m = wave * 16 + ((lane < 16) ? r : r + 8);
            long orow = tile + m;
            if (orow < N_NODES) out[orow * DIM + n] = acc[nt][r] + b;
        }
    }
}

extern "C" void kernel_launch(void* const* d_in, const int* in_sizes, int n_in,
                              void* d_out, int out_size, void* d_ws, size_t ws_size,
                              hipStream_t stream) {
    const float* node_feat = (const float*)d_in[0];
    const float* edge_feat = (const float*)d_in[1];
    const int*   src_idx   = (const int*)d_in[2];
    const int*   dst_idx   = (const int*)d_in[3];
    const float* Wsrc  = (const float*)d_in[4];  const float* bsrc  = (const float*)d_in[5];
    const float* Wdst  = (const float*)d_in[6];  const float* bdst  = (const float*)d_in[7];
    const float* Wphi1 = (const float*)d_in[8];  const float* bphi1 = (const float*)d_in[9];
    const float* Wphi2 = (const float*)d_in[10]; const float* bphi2 = (const float*)d_in[11];
    const float* Wphi3 = (const float*)d_in[12]; const float* bphi3 = (const float*)d_in[13];
    const float* Wth1  = (const float*)d_in[14]; const float* bth1  = (const float*)d_in[15];
    const float* Wth2  = (const float*)d_in[16]; const float* bth2  = (const float*)d_in[17];
    const float* Wpd   = (const float*)d_in[18]; const float* bpd   = (const float*)d_in[19];
    const float* Wpu   = (const float*)d_in[20]; const float* bpu   = (const float*)d_in[21];

    float*    agg   = (float*)d_ws;
    unsigned* wpack = (unsigned*)((char*)d_ws + (size_t)N_NODES * DIM * sizeof(float));

    (void)hipMemsetAsync(agg, 0, (size_t)N_NODES * DIM * sizeof(float), stream);

    const float* Ws[9] = {Wsrc, Wdst, Wphi1, Wphi2, Wphi3, Wth1, Wth2, Wpd, Wpu};
    for (int i = 0; i < 9; ++i)
        pack_w_kernel<<<1, 256, 0, stream>>>(Ws[i], wpack + (size_t)i * 8192);

    gnn_edge_kernel<<<N_EDGES / TILE, 256, 0, stream>>>(
        node_feat, edge_feat, src_idx, dst_idx, wpack,
        bsrc, bdst, bphi1, bphi2, bphi3, agg);

    gnn_node_kernel<<<(N_NODES + TILE - 1) / TILE, 256, 0, stream>>>(
        node_feat, agg, wpack, bpd, bpu, bth1, bth2, (float*)d_out);
}
